// RNNAgent_52063593562852
// MI455X (gfx1250) — matrix operand order repaired and checked by reading, not compiled, not run on hardware
//
#include <hip/hip_runtime.h>
#include <hip/hip_bf16.h>

typedef __attribute__((ext_vector_type(16))) _Float16 v16h;
typedef __attribute__((ext_vector_type(8)))  _Float16 v8h;
typedef __attribute__((ext_vector_type(8)))  float    v8f;

#define BB 16384
#define AA 16
#define FF 96
#define H2 256
#define NAA 16
#define NN (BB * (AA - 1))   /* 245760 */
#define F2 192
#define FINAL_ELEMS (BB * AA * NAA)   /* 4194304 */

/* ---------- workspace layout (bytes) ---------- */
#define OFF_XH    ((size_t)0)           /* N*256 f16 = 125829120 */
#define OFF_HPH   ((size_t)125829120)   /* N*256 f16 = 125829120 */
#define OFF_HH16  ((size_t)251658240)   /* N*256 f16 = 125829120 */
#define OFF_DEC   ((size_t)377487360)   /* 2N*16 f32 = 31457280  */
#define OFF_W1T   ((size_t)408944640)   /* 256*192 f16 */
#define OFF_WIHT  ((size_t)409042944)   /* 768*256 f16 */
#define OFF_WHHT  ((size_t)409436160)   /* 768*256 f16 */
#define OFF_W2T   ((size_t)409829376)   /* 256*256 f16 */
#define OFF_MID   ((size_t)409960448)   /* 14*16 f32 */

__device__ __forceinline__ v8f wmma16(v16h a, v16h b, v8f c) {
  return __builtin_amdgcn_wmma_f32_16x16x32_f16(false, a, false, b, (short)0, c,
                                                false, false);
}

__device__ __forceinline__ v16h cat8(v8h lo, v8h hi) {
  return __builtin_shufflevector(lo, hi, 0, 1, 2, 3, 4, 5, 6, 7, 8, 9, 10, 11,
                                 12, 13, 14, 15);
}

/* A fragment (16x32 f16): lane&15 = row M; lane>>4 selects K offset 0/8;
   VGPRs 0-3 hold K=koff..koff+7, VGPRs 4-7 hold K=16+koff..16+koff+7. */
__device__ __forceinline__ v16h a_frag_f16(const _Float16* __restrict__ rowp,
                                           int k0, int lane) {
  int koff = (lane >> 4) << 3;
  const v8h* p = (const v8h*)(rowp + k0 + koff);
  return cat8(p[0], p[2]);
}

/* same but source is f32 (convert inline) */
__device__ __forceinline__ v16h a_frag_f32(const float* __restrict__ rowp,
                                           int k0, int lane) {
  int koff = (lane >> 4) << 3;
  const float* p = rowp + k0 + koff;
  v16h a;
#pragma unroll
  for (int i = 0; i < 8; ++i) {
    a[i] = (_Float16)p[i];
    a[i + 8] = (_Float16)p[i + 16];
  }
  return a;
}

/* B fragment (32x16 f16) from N-major weights Wt[Nout][K]: lane&15 = column N,
   lane>>4 selects K offset 0/8; elements 0-7 K=koff.., 8-15 K=16+koff... */
__device__ __forceinline__ v16h b_frag(const _Float16* __restrict__ wt, int K,
                                       int col0, int k0, int lane) {
  const _Float16* p =
      wt + (size_t)(col0 + (lane & 15)) * K + k0 + ((lane >> 4) << 3);
  const v8h* q = (const v8h*)p;
  return cat8(q[0], q[2]);
}

/* B fragment from LDS-staged tile: p0 points at [16][256] f16 slab */
__device__ __forceinline__ v16h b_frag_sh(const _Float16* p0, int k0, int lane) {
  const _Float16* p = p0 + ((lane & 15) << 8) + k0 + ((lane >> 4) << 3);
  const v8h* q = (const v8h*)p;
  return cat8(q[0], q[2]);
}

/* ---------- prep: f32 KxN row-major -> f16 NxK (N-major) ---------- */
__global__ void k_cvt_t(const float* __restrict__ in, _Float16* __restrict__ out,
                        int K, int Nout) {
  int idx = blockIdx.x * 256 + threadIdx.x;
  if (idx < K * Nout) {
    int n = idx / K;
    int k = idx - n * K;
    out[idx] = (_Float16)in[(size_t)k * Nout + n];
  }
}

/* ---------- prep: f32 -> f16 elementwise (4/thread) ---------- */
__global__ void k_cvt(const float* __restrict__ in, _Float16* __restrict__ out,
                      int n4) {
  int i = (blockIdx.x * 256 + threadIdx.x) * 4;
  if (i < n4) {
#pragma unroll
    for (int j = 0; j < 4; ++j) out[i + j] = (_Float16)in[i + j];
  }
}

/* ---------- GEMM1: xh = relu(X @ w1 + b1); 32-row tile per wave ---------- */
__global__ void __launch_bounds__(256)
k_gemm1(const float* __restrict__ inputs, const _Float16* __restrict__ w1t,
        const float* __restrict__ bias1, _Float16* __restrict__ xh) {
  int lane = threadIdx.x & 31;
  int tile = blockIdx.x * 8 + (threadIdx.x >> 5);
  int rowTile2 = tile >> 4;
  int colTile = tile & 15;
  int row0 = rowTile2 * 32;

  /* X row n is the contiguous 192-float window inputs + (b*16+t)*96 */
  int n0 = row0 + (lane & 15);
  int n1 = n0 + 16;
  int q0 = n0 / 15, t0 = n0 - q0 * 15;
  int q1 = n1 / 15, t1 = n1 - q1 * 15;
  const float* rowp0 = inputs + (size_t)(q0 * 16 + t0) * FF;
  const float* rowp1 = inputs + (size_t)(q1 * 16 + t1) * FF;

  v8f acc0 = {}, acc1 = {};
#pragma unroll
  for (int ks = 0; ks < 6; ++ks) {
    v16h w = b_frag(w1t, F2, colTile * 16, ks * 32, lane);
    v16h a0 = a_frag_f32(rowp0, ks * 32, lane);
    acc0 = wmma16(a0, w, acc0);
    v16h a1 = a_frag_f32(rowp1, ks * 32, lane);
    acc1 = wmma16(a1, w, acc1);
  }
  int colg = colTile * 16 + (lane & 15);
  float bias = bias1[colg];
  int rbase = row0 + ((lane >> 4) << 3);
#pragma unroll
  for (int e = 0; e < 8; ++e) {
    float v0 = acc0[e] + bias;
    float v1 = acc1[e] + bias;
    xh[(size_t)(rbase + e) * H2 + colg] = (_Float16)(v0 > 0.f ? v0 : 0.f);
    xh[(size_t)(rbase + 16 + e) * H2 + colg] = (_Float16)(v1 > 0.f ? v1 : 0.f);
  }
}

/* ---------- GEMM2: gi/gh (3 gates each) + fused GRU -> hp ----------
   Block = 128 rows x 16 cols; all 6 gate B-tiles (48KB) staged in LDS and
   shared by 8 waves (8x less weight traffic from L2). */
__global__ void __launch_bounds__(256)
k_gemm2(const _Float16* __restrict__ xh, const _Float16* __restrict__ hh16,
        const float* __restrict__ hs,
        const _Float16* __restrict__ wiht, const _Float16* __restrict__ whht,
        const float* __restrict__ bih, const float* __restrict__ bhh,
        float* __restrict__ hp_out, _Float16* __restrict__ hp_h) {
  __shared__ _Float16 bsh[6 * 16 * H2]; /* [gate][col16][K256] = 48KB */
  int lane = threadIdx.x & 31;
  int wv = threadIdx.x >> 5;
  int colTile = blockIdx.x & 15;
  int rowBlock = blockIdx.x >> 4;

  /* stage B: 3072 v8h chunks across 256 threads (12 iters, uniform) */
#pragma unroll
  for (int it = 0; it < 12; ++it) {
    int i = it * 256 + threadIdx.x;
    int g = i >> 9;          /* 512 chunks per gate-matrix */
    int r = i & 511;
    int c = r >> 5;
    int k8 = (r & 31) << 3;
    int gg = g < 3 ? g : g - 3;
    const _Float16* src =
        (g < 3 ? wiht : whht) + (size_t)(gg * H2 + colTile * 16 + c) * H2 + k8;
    *(v8h*)(bsh + (size_t)i * 8) = *(const v8h*)src;
  }
  __syncthreads();

  int row0 = (rowBlock * 8 + wv) * 16;
  const _Float16* xrowp = xh + (size_t)(row0 + (lane & 15)) * H2;
  const _Float16* hrowp = hh16 + (size_t)(row0 + (lane & 15)) * H2;

  v8f aci[3] = {{}, {}, {}};
  v8f ach[3] = {{}, {}, {}};
#pragma unroll
  for (int ks = 0; ks < 8; ++ks) {
    v16h ax = a_frag_f16(xrowp, ks * 32, lane);
    v16h ah = a_frag_f16(hrowp, ks * 32, lane);
#pragma unroll
    for (int g = 0; g < 3; ++g) {
      v16h bi = b_frag_sh(bsh + g * (16 * H2), ks * 32, lane);
      aci[g] = wmma16(ax, bi, aci[g]);
      v16h bh = b_frag_sh(bsh + (3 + g) * (16 * H2), ks * 32, lane);
      ach[g] = wmma16(ah, bh, ach[g]);
    }
  }

  int colg = colTile * 16 + (lane & 15);
  float bir = bih[colg], biz = bih[H2 + colg], bin_ = bih[2 * H2 + colg];
  float bhr = bhh[colg], bhz = bhh[H2 + colg], bhn = bhh[2 * H2 + colg];
  int rbase = row0 + ((lane >> 4) << 3);
#pragma unroll
  for (int e = 0; e < 8; ++e) {
    float gr = (aci[0][e] + bir) + (ach[0][e] + bhr);
    float gz = (aci[1][e] + biz) + (ach[1][e] + bhz);
    float r = 1.f / (1.f + __expf(-gr));
    float z = 1.f / (1.f + __expf(-gz));
    float nn = tanhf((aci[2][e] + bin_) + r * (ach[2][e] + bhn));
    size_t idx = (size_t)(rbase + e) * H2 + colg;
    float h = hs[idx];
    float hpv = (1.f - z) * nn + z * h;
    hp_out[idx] = hpv;
    hp_h[idx] = (_Float16)hpv;
  }
}

/* ---------- GEMM3: qp = hp @ w2 + b2, fused row/col maxes -> dec ----------
   Block = 64 rows x 256 cols; wave owns one column tile across 4 row
   subtiles so each B fragment is reused 4x. */
__global__ void __launch_bounds__(512)
k_gemm3(const _Float16* __restrict__ hp_h, const _Float16* __restrict__ w2t,
        const float* __restrict__ b2, float* __restrict__ dec) {
  __shared__ float qp[64 * 256]; /* 64KB */
  int lane = threadIdx.x & 31;
  int wv = threadIdx.x >> 5; /* 0..15 -> column tile */
  int base = blockIdx.x * 64;

  const _Float16* arow[4];
#pragma unroll
  for (int s = 0; s < 4; ++s)
    arow[s] = hp_h + (size_t)(base + s * 16 + (lane & 15)) * H2;

  v8f acc[4] = {{}, {}, {}, {}};
#pragma unroll
  for (int ks = 0; ks < 8; ++ks) {
    v16h w = b_frag(w2t, H2, wv * 16, ks * 32, lane);
#pragma unroll
    for (int s = 0; s < 4; ++s) {
      v16h a = a_frag_f16(arow[s], ks * 32, lane);
      acc[s] = wmma16(a, w, acc[s]);
    }
  }
  int col = wv * 16 + (lane & 15);
  float bias = b2[col];
  int rb = (lane >> 4) << 3;
#pragma unroll
  for (int s = 0; s < 4; ++s)
#pragma unroll
    for (int e = 0; e < 8; ++e)
      qp[(s * 16 + rb + e) * 256 + col] = acc[s][e] + bias;
  __syncthreads();

#pragma unroll
  for (int it = 0; it < 2; ++it) {
    int rr = it * 512 + threadIdx.x; /* 1024 (row,k) jobs */
    int row = rr >> 4;
    int k = rr & 15;
    float lmax = -3.4e38f, rmax = -3.4e38f;
#pragma unroll
    for (int j = 0; j < 16; ++j) {
      lmax = fmaxf(lmax, qp[row * 256 + k * 16 + j]); /* max over axis=2 */
      rmax = fmaxf(rmax, qp[row * 256 + j * 16 + k]); /* max over axis=1 */
    }
    int n = base + row;
    int hi = k >> 3, kk = k & 7;
    dec[(size_t)(2 * n + hi) * 16 + 2 * kk] = lmax;
    dec[(size_t)(2 * n + hi) * 16 + 2 * kk + 1] = rmax;
  }
}

/* ---------- mid: 14x16 decision from dec rows 1..28 ---------- */
__global__ void k_mid(const float* __restrict__ dec, float* __restrict__ mid) {
  __shared__ float sL[16][16], sR[16][16];
  int i = threadIdx.x >> 4, c = threadIdx.x & 15;
  float Lv = 0.f, Rv = 0.f;
  if (i >= 1 && i <= 14) {
    Lv = dec[i * 16 + c];        /* partial[0, i, c]   */
    Rv = dec[(14 + i) * 16 + c]; /* partial[1, i-1, c] */
  }
  sL[i][c] = Lv;
  sR[i][c] = Rv;
  __syncthreads();
  if (i >= 1 && i <= 14) {
    float ml = -3.4e38f, mr = -3.4e38f;
#pragma unroll
    for (int j = 0; j < 16; ++j) {
      ml = fmaxf(ml, sL[i][j]);
      mr = fmaxf(mr, sR[i][j]);
    }
    mid[(i - 1) * 16 + c] = (ml > mr) ? Lv : Rv;
  }
}

/* ---------- final assembly: (B, 16, 16) ---------- */
__global__ void k_final(const float* __restrict__ dec,
                        const float* __restrict__ mid,
                        float* __restrict__ out) {
  int b = blockIdx.x;
  int t = threadIdx.x;
  int a = t >> 4, c = t & 15;
  float v;
  if (a == 0)
    v = dec[(size_t)(b * 15) * 16 + c];
  else if (a == 15)
    v = dec[(size_t)(b * 15 + 14) * 16 + c];
  else
    v = mid[(a - 1) * 16 + c];
  out[(size_t)b * 256 + t] = v;
}

extern "C" void kernel_launch(void* const* d_in, const int* in_sizes, int n_in,
                              void* d_out, int out_size, void* d_ws,
                              size_t ws_size, hipStream_t stream) {
  (void)in_sizes; (void)n_in; (void)out_size; (void)ws_size;
  const float* inputs = (const float*)d_in[0];
  const float* hidden = (const float*)d_in[1];
  const float* w1 = (const float*)d_in[2];
  const float* b1 = (const float*)d_in[3];
  const float* w_ih = (const float*)d_in[4];
  const float* b_ih = (const float*)d_in[5];
  const float* w_hh = (const float*)d_in[6];
  const float* b_hh = (const float*)d_in[7];
  const float* w2 = (const float*)d_in[8];
  const float* b2 = (const float*)d_in[9];

  char* ws = (char*)d_ws;
  _Float16* xh = (_Float16*)(ws + OFF_XH);
  _Float16* hph = (_Float16*)(ws + OFF_HPH);
  _Float16* hh16 = (_Float16*)(ws + OFF_HH16);
  float* dec = (float*)(ws + OFF_DEC);
  _Float16* w1t = (_Float16*)(ws + OFF_W1T);
  _Float16* wiht = (_Float16*)(ws + OFF_WIHT);
  _Float16* whht = (_Float16*)(ws + OFF_WHHT);
  _Float16* w2t = (_Float16*)(ws + OFF_W2T);
  float* mid = (float*)(ws + OFF_MID);

  float* out_final = (float*)d_out;
  float* out_hp = (float*)d_out + FINAL_ELEMS;

  /* weight convert + transpose (tiny; resident in L2 afterwards) */
  k_cvt_t<<<(F2 * H2 + 255) / 256, 256, 0, stream>>>(w1, w1t, F2, H2);
  k_cvt_t<<<(H2 * 3 * H2 + 255) / 256, 256, 0, stream>>>(w_ih, wiht, H2, 3 * H2);
  k_cvt_t<<<(H2 * 3 * H2 + 255) / 256, 256, 0, stream>>>(w_hh, whht, H2, 3 * H2);
  k_cvt_t<<<(H2 * H2 + 255) / 256, 256, 0, stream>>>(w2, w2t, H2, H2);
  /* hidden f32 -> f16 once (keeps GEMM2 inner loop cvt-free) */
  k_cvt<<<(NN * H2 / 4 + 255) / 256, 256, 0, stream>>>(hidden, hh16, NN * H2);

  k_gemm1<<<(NN / 32) * 16 / 8, 256, 0, stream>>>(inputs, w1t, b1, xh);
  k_gemm2<<<(NN / 16 / 8) * 16, 256, 0, stream>>>(xh, hh16, hidden, wiht, whht,
                                                  b_ih, b_hh, out_hp, hph);
  k_gemm3<<<NN / 64, 512, 0, stream>>>(hph, w2t, b2, dec);
  k_mid<<<1, 256, 0, stream>>>(dec, mid);
  k_final<<<BB, 256, 0, stream>>>(dec, mid, out_final);
}
